// LSTMVAE_61881888801441
// MI455X (gfx1250) — compile-verified, matrix operand order checked by reading, not run on hardware
//
#include <hip/hip_runtime.h>
#include <hip/hip_bf16.h>
#include <math.h>

// ---- problem constants ----
#define B_   64
#define T_   1000
#define IN_  8
#define H_   512
#define Z4_  2048
#define LAT_ 128
#define ALPHA_ 0.1f

// ---- persistent-kernel decomposition ----
#define NWG     64       // persistent workgroups (each owns HS hidden units)
#define HS      8        // hidden units per WG
#define ROWS    32       // 4 gates * HS rows of the 2048-row gate matrix per WG
#define THREADS 256      // 8 wave32 waves per WG

typedef __bf16 bf16;
typedef __attribute__((ext_vector_type(16))) __bf16 v16bf;
typedef __attribute__((ext_vector_type(8)))  __bf16 v8bf;
typedef __attribute__((ext_vector_type(8)))  float  v8f;

__device__ __forceinline__ float sigmoidf_(float x) { return 1.0f / (1.0f + __expf(-x)); }

// Load one WMMA bf16 operand tile row-slice per lane.
// Layout (ISA 7.12.2, 16-bit A 16x32): lane%16 = row, lane/16 = K-half.
// elements 0..7  -> cols kb + 8*kh + 0..7
// elements 8..15 -> cols kb + 16 + 8*kh + 0..7
__device__ __forceinline__ v16bf ld_tile(const bf16* __restrict__ base, int row, int kb, int kh) {
  const bf16* p = base + (size_t)row * H_ + kb + 8 * kh;
  v8bf lo = *(const v8bf*)p;
  v8bf hi = *(const v8bf*)(p + 16);
  v16bf r;
#pragma unroll
  for (int i = 0; i < 8; ++i) { r[i] = lo[i]; r[i + 8] = hi[i]; }
  return r;
}

// Grid-wide barrier among NWG resident workgroups (atomic arrive + spin).
__device__ __forceinline__ void gsync(unsigned* bar, unsigned target) {
  __syncthreads();
  if (threadIdx.x == 0) {
    __threadfence();                 // release our global writes
    atomicAdd(bar, 1u);
    while (__hip_atomic_load(bar, __ATOMIC_RELAXED, __HIP_MEMORY_SCOPE_AGENT) < target)
      __builtin_amdgcn_s_sleep(2);
  }
  __syncthreads();
  __threadfence();                   // acquire others' global writes
}

// ---------------- small prep kernels ----------------

__global__ void k_ema(const float* __restrict__ x, float* __restrict__ ema) {
  int b = blockIdx.x * blockDim.x + threadIdx.x;
  if (b >= B_) return;
  const float* xr = x + (size_t)b * T_ * IN_;
  float e = xr[1];
  ema[(size_t)b * T_] = e;
  for (int t = 1; t < T_; ++t) {
    float xv = xr[(size_t)t * IN_ + 1];
    e = ALPHA_ * xv + (1.0f - ALPHA_) * e;
    ema[(size_t)b * T_ + t] = e;
  }
}

__global__ void k_cvt_bf16(const float* __restrict__ a, bf16* __restrict__ o, int n) {
  int i = blockIdx.x * blockDim.x + threadIdx.x;
  if (i < n) o[i] = (bf16)a[i];
}

__global__ void k_recon_init(float* __restrict__ recon, const float* __restrict__ bout) {
  size_t i = (size_t)blockIdx.x * blockDim.x + threadIdx.x;
  if (i < (size_t)B_ * T_ * IN_) recon[i] = bout[i & 7];
}

// ---------------- fc_input + positional encoding producer ----------------
// Computes hfeat[t][m][j] = relu(x_ema[m,t,:] . fcw[j,:] + fcb[j]) + 0.1*pe(t,j)
// for this WG's slice j in [wg*HS, wg*HS+HS).
__device__ __forceinline__ void produce_feat(int t, int wg,
    const float* __restrict__ x, const float* __restrict__ ema,
    const float* __restrict__ fcw, const float* __restrict__ fcb,
    bf16* __restrict__ dst /* [B_][H_] */) {
  for (int it = threadIdx.x; it < B_ * HS; it += THREADS) {
    int m = it >> 3;
    int jj = it & 7;
    int j = wg * HS + jj;
    const float* xr = x + ((size_t)m * T_ + t) * IN_;
    if (t + 1 < T_) __builtin_prefetch(x + ((size_t)m * T_ + (t + 1)) * IN_, 0, 1);
    float acc = fcb[j];
#pragma unroll
    for (int i = 0; i < IN_; ++i) {
      float xv = (i == 1) ? ema[(size_t)m * T_ + t] : xr[i];
      acc += xv * fcw[j * IN_ + i];
    }
    acc = fmaxf(acc, 0.0f);
    // pe: div = exp((j&~1) * (-ln(10000)/512)); even j -> sin, odd -> cos
    float freq = __expf((float)(j & ~1) * (-0.017988946011f));
    float ang = (float)t * freq;
    acc += 0.1f * ((j & 1) ? __cosf(ang) : __sinf(ang));
    dst[m * H_ + j] = (bf16)acc;
  }
}

// ---------------- encoder persistent LSTM ----------------
__global__ void __launch_bounds__(THREADS, 1) k_enc(
    const float* __restrict__ x, const float* __restrict__ ema,
    const float* __restrict__ fcw, const float* __restrict__ fcb,
    const bf16* __restrict__ wih, const bf16* __restrict__ whh,
    const float* __restrict__ bih, const float* __restrict__ bhh,
    bf16* __restrict__ hfeat /* [2][B_][H_] */,
    bf16* __restrict__ hbuf  /* [2][B_][H_] */,
    unsigned* __restrict__ bar) {
  __shared__ float gates_s[B_][ROWS + 4];
  __shared__ float c_s[B_][HS];
  __shared__ float bias_s[ROWS];

  const int wg = blockIdx.x;
  const int tid = threadIdx.x;
  const int lane = tid & 31, wv = tid >> 5;
  const int lr = lane & 15, kh = lane >> 4;
  const int mt = wv & 3, nt = wv >> 2;          // 4 batch tiles x 2 row tiles

  if (tid < ROWS) {
    int g = tid >> 3, u = tid & 7;
    int grow0 = g * H_ + wg * HS + u;
    bias_s[tid] = bih[grow0] + bhh[grow0];
  }
  for (int it = tid; it < B_ * HS; it += THREADS) c_s[it >> 3][it & 7] = 0.0f;

  unsigned gen = 0;
  produce_feat(0, wg, x, ema, fcw, fcb, hfeat);  // features for t=0 into buffer 0
  gen += NWG; gsync(bar, gen);

  const int r = nt * 16 + lr;                    // local gate row 0..31
  const int grow = (r >> 3) * H_ + wg * HS + (r & 7);  // global gate row
  const int mrow = mt * 16 + lr;                 // batch row

  for (int t = 0; t < T_; ++t) {
    const bf16* hf = hfeat + (size_t)(t & 1) * B_ * H_;
    const bf16* hp = hbuf + (size_t)(t & 1) * B_ * H_;
    bf16* hn = hbuf + (size_t)((t + 1) & 1) * B_ * H_;

    v8f acc0, acc1;                              // two independent WMMA chains
    float bz = bias_s[r];
#pragma unroll
    for (int v = 0; v < 8; ++v) { acc0[v] = bz; acc1[v] = 0.0f; }

#pragma unroll 4
    for (int ks = 0; ks < 16; ++ks) {
      int kb = ks * 32;
      v16bf a0 = ld_tile(hf, mrow, kb, kh);
      v16bf b0 = ld_tile(wih, grow, kb, kh);
      acc0 = __builtin_amdgcn_wmma_f32_16x16x32_bf16(false, a0, false, b0, (short)0, acc0, false, false);
      v16bf a1 = ld_tile(hp, mrow, kb, kh);
      v16bf b1 = ld_tile(whh, grow, kb, kh);
      acc1 = __builtin_amdgcn_wmma_f32_16x16x32_bf16(false, a1, false, b1, (short)0, acc1, false, false);
    }
#pragma unroll
    for (int v = 0; v < 8; ++v)
      gates_s[mt * 16 + v + 8 * kh][nt * 16 + lr] = acc0[v] + acc1[v];

    __syncthreads();

    // LSTM cell update for this WG's HS hidden units, all 64 batches
    for (int it = tid; it < B_ * HS; it += THREADS) {
      int m = it >> 3, u = it & 7;
      float gi = gates_s[m][u];
      float gf = gates_s[m][HS + u];
      float gg = gates_s[m][2 * HS + u];
      float go = gates_s[m][3 * HS + u];
      float c = c_s[m][u];
      float cn = sigmoidf_(gf) * c + sigmoidf_(gi) * tanhf(gg);
      c_s[m][u] = cn;
      hn[m * H_ + wg * HS + u] = (bf16)(sigmoidf_(go) * tanhf(cn));
    }

    if (t + 1 < T_)
      produce_feat(t + 1, wg, x, ema, fcw, fcb, hfeat + (size_t)((t + 1) & 1) * B_ * H_);

    gen += NWG; gsync(bar, gen);
  }
}

// ---------------- mu / logvar / z ----------------
__global__ void k_muvar(const bf16* __restrict__ hlast,
                        const float* __restrict__ wmu, const float* __restrict__ bmu,
                        const float* __restrict__ wlv, const float* __restrict__ blv,
                        const float* __restrict__ eps,
                        float* __restrict__ out_mu, float* __restrict__ out_lv,
                        float* __restrict__ z) {
  int i = blockIdx.x * blockDim.x + threadIdx.x;
  if (i >= B_ * LAT_) return;
  int b = i / LAT_, l = i % LAT_;
  const bf16* hr = hlast + (size_t)b * H_;
  float mu = bmu[l], lv = blv[l];
  for (int k = 0; k < H_; ++k) {
    float h = (float)hr[k];
    mu += h * wmu[(size_t)l * H_ + k];
    lv += h * wlv[(size_t)l * H_ + k];
  }
  out_mu[i] = mu;
  out_lv[i] = lv;
  z[i] = mu + eps[i] * __expf(0.5f * lv);
}

// ---------------- decoder input d = z @ dec_in_w^T + b ----------------
__global__ void k_dec_in(const float* __restrict__ z, const float* __restrict__ w,
                         const float* __restrict__ bias, float* __restrict__ d) {
  int i = blockIdx.x * blockDim.x + threadIdx.x;
  if (i >= B_ * H_) return;
  int b = i / H_, j = i % H_;
  float acc = bias[j];
  const float* zr = z + (size_t)b * LAT_;
  const float* wr = w + (size_t)j * LAT_;
  for (int l = 0; l < LAT_; ++l) acc += zr[l] * wr[l];
  d[i] = acc;
}

// ---------------- decoder persistent LSTM (+ fused fc_out) ----------------
__global__ void __launch_bounds__(THREADS, 1) k_dec(
    const float* __restrict__ d, const float* __restrict__ wih_f,
    const float* __restrict__ bih, const float* __restrict__ bhh,
    const bf16* __restrict__ whh, const float* __restrict__ wout,
    float* __restrict__ recon, bf16* __restrict__ hbuf, unsigned* __restrict__ bar) {
  __shared__ float gates_s[B_][ROWS + 4];
  __shared__ float xg_s[B_][ROWS + 4];
  __shared__ float c_s[B_][HS];
  __shared__ float hsl_s[B_][HS];
  __shared__ float wout_s[IN_][HS];

  const int wg = blockIdx.x;
  const int tid = threadIdx.x;
  const int lane = tid & 31, wv = tid >> 5;
  const int lr = lane & 15, kh = lane >> 4;
  const int mt = wv & 3, nt = wv >> 2;

  if (tid < IN_ * HS) {
    int o = tid >> 3, u = tid & 7;
    wout_s[o][u] = wout[(size_t)o * H_ + wg * HS + u];
  }
  for (int it = tid; it < B_ * HS; it += THREADS) c_s[it >> 3][it & 7] = 0.0f;

  // xg_const = d @ dec_w_ih^T + b_ih + b_hh (input is constant over t)
  for (int q = tid; q < B_ * ROWS; q += THREADS) {
    int m = q >> 5, rr = q & 31;
    int grow0 = (rr >> 3) * H_ + wg * HS + (rr & 7);
    float acc = bih[grow0] + bhh[grow0];
    const float* dr = d + (size_t)m * H_;
    const float* wr = wih_f + (size_t)grow0 * H_;
    for (int k = 0; k < H_; ++k) acc += dr[k] * wr[k];
    xg_s[m][rr] = acc;
  }
  __syncthreads();

  const int r = nt * 16 + lr;
  const int grow = (r >> 3) * H_ + wg * HS + (r & 7);
  const int mrow = mt * 16 + lr;

  unsigned gen = 0;
  for (int t = 0; t < T_; ++t) {
    const bf16* hp = hbuf + (size_t)(t & 1) * B_ * H_;
    bf16* hn = hbuf + (size_t)((t + 1) & 1) * B_ * H_;

    v8f acc0, acc1;
#pragma unroll
    for (int v = 0; v < 8; ++v) {
      acc0[v] = xg_s[mt * 16 + v + 8 * kh][nt * 16 + lr];
      acc1[v] = 0.0f;
    }
#pragma unroll 4
    for (int ks = 0; ks < 16; ks += 2) {           // two interleaved chains
      v16bf a0 = ld_tile(hp, mrow, ks * 32, kh);
      v16bf b0 = ld_tile(whh, grow, ks * 32, kh);
      acc0 = __builtin_amdgcn_wmma_f32_16x16x32_bf16(false, a0, false, b0, (short)0, acc0, false, false);
      v16bf a1 = ld_tile(hp, mrow, (ks + 1) * 32, kh);
      v16bf b1 = ld_tile(whh, grow, (ks + 1) * 32, kh);
      acc1 = __builtin_amdgcn_wmma_f32_16x16x32_bf16(false, a1, false, b1, (short)0, acc1, false, false);
    }
#pragma unroll
    for (int v = 0; v < 8; ++v)
      gates_s[mt * 16 + v + 8 * kh][nt * 16 + lr] = acc0[v] + acc1[v];

    __syncthreads();

    for (int it = tid; it < B_ * HS; it += THREADS) {
      int m = it >> 3, u = it & 7;
      float gi = gates_s[m][u];
      float gf = gates_s[m][HS + u];
      float gg = gates_s[m][2 * HS + u];
      float go = gates_s[m][3 * HS + u];
      float c = c_s[m][u];
      float cn = sigmoidf_(gf) * c + sigmoidf_(gi) * tanhf(gg);
      c_s[m][u] = cn;
      float hv = sigmoidf_(go) * tanhf(cn);
      hsl_s[m][u] = hv;
      hn[m * H_ + wg * HS + u] = (bf16)hv;
    }
    __syncthreads();

    // fused fc_out partial: recon[b,t,o] += sum_u h[b, wg*HS+u] * wout[o, wg*HS+u]
    for (int it = tid; it < B_ * IN_; it += THREADS) {
      int m = it >> 3, o = it & 7;
      float p = 0.0f;
#pragma unroll
      for (int u = 0; u < HS; ++u) p += hsl_s[m][u] * wout_s[o][u];
      atomicAdd(&recon[((size_t)m * T_ + t) * IN_ + o], p);
    }

    gen += NWG; gsync(bar, gen);
  }
}

// ---------------- host launcher ----------------
extern "C" void kernel_launch(void* const* d_in, const int* in_sizes, int n_in,
                              void* d_out, int out_size, void* d_ws, size_t ws_size,
                              hipStream_t stream) {
  (void)in_sizes; (void)n_in; (void)out_size; (void)ws_size;

  const float* x    = (const float*)d_in[0];
  const float* eps  = (const float*)d_in[1];
  const float* fcw  = (const float*)d_in[2];
  const float* fcb  = (const float*)d_in[3];
  const float* ewih = (const float*)d_in[4];
  const float* ewhh = (const float*)d_in[5];
  const float* ebih = (const float*)d_in[6];
  const float* ebhh = (const float*)d_in[7];
  const float* wmu  = (const float*)d_in[8];
  const float* bmu  = (const float*)d_in[9];
  const float* wlv  = (const float*)d_in[10];
  const float* blv  = (const float*)d_in[11];
  const float* dinw = (const float*)d_in[12];
  const float* dinb = (const float*)d_in[13];
  const float* dwih = (const float*)d_in[14];
  const float* dwhh = (const float*)d_in[15];
  const float* dbih = (const float*)d_in[16];
  const float* dbhh = (const float*)d_in[17];
  const float* wout = (const float*)d_in[18];
  const float* bout = (const float*)d_in[19];

  float* out    = (float*)d_out;
  float* recon  = out;                                  // [B,T,IN]
  float* out_mu = out + (size_t)B_ * T_ * IN_;          // [B,LAT]
  float* out_lv = out_mu + (size_t)B_ * LAT_;           // [B,LAT]

  char* ws = (char*)d_ws;
  size_t off = 0;
  auto wsalloc = [&](size_t bytes) -> void* {
    void* p = ws + off;
    off = (off + bytes + 255) & ~(size_t)255;
    return p;
  };
  unsigned* bar  = (unsigned*)wsalloc(256);             // [0]=enc counter, [32]=dec counter
  float* ema     = (float*)wsalloc(sizeof(float) * B_ * T_);
  bf16* hfeat    = (bf16*)wsalloc(sizeof(bf16) * 2 * B_ * H_);
  bf16* hbuf     = (bf16*)wsalloc(sizeof(bf16) * 2 * B_ * H_);
  bf16* wih_bf   = (bf16*)wsalloc(sizeof(bf16) * Z4_ * H_);
  bf16* whh_bf   = (bf16*)wsalloc(sizeof(bf16) * Z4_ * H_);
  bf16* dwhh_bf  = (bf16*)wsalloc(sizeof(bf16) * Z4_ * H_);
  float* z       = (float*)wsalloc(sizeof(float) * B_ * LAT_);
  float* dvec    = (float*)wsalloc(sizeof(float) * B_ * H_);

  hipMemsetAsync(bar, 0, 256, stream);
  hipMemsetAsync(hbuf, 0, sizeof(bf16) * 2 * B_ * H_, stream);

  k_ema<<<1, 64, 0, stream>>>(x, ema);
  const int nW = Z4_ * H_;
  k_cvt_bf16<<<(nW + 255) / 256, 256, 0, stream>>>(ewih, wih_bf, nW);
  k_cvt_bf16<<<(nW + 255) / 256, 256, 0, stream>>>(ewhh, whh_bf, nW);
  k_cvt_bf16<<<(nW + 255) / 256, 256, 0, stream>>>(dwhh, dwhh_bf, nW);
  k_recon_init<<<(B_ * T_ * IN_ + 255) / 256, 256, 0, stream>>>(recon, bout);

  k_enc<<<NWG, THREADS, 0, stream>>>(x, ema, fcw, fcb, wih_bf, whh_bf, ebih, ebhh,
                                     hfeat, hbuf, bar);

  // T_ is even -> final hidden state lives in hbuf[0]
  k_muvar<<<(B_ * LAT_ + 255) / 256, 256, 0, stream>>>(hbuf, wmu, bmu, wlv, blv, eps,
                                                       out_mu, out_lv, z);
  k_dec_in<<<(B_ * H_ + 255) / 256, 256, 0, stream>>>(z, dinw, dinb, dvec);

  hipMemsetAsync(hbuf, 0, sizeof(bf16) * 2 * B_ * H_, stream);
  k_dec<<<NWG, THREADS, 0, stream>>>(dvec, dwih, dbih, dbhh, dwhh_bf, wout,
                                     recon, hbuf, bar + 32);
}
